// VectorQuantizer_14199161880608
// MI455X (gfx1250) — matrix operand order repaired and checked by reading, not compiled, not run on hardware
//
#include <hip/hip_runtime.h>

// VQ argmin via WMMA bf16 (hi/lo split for fp32-grade accuracy).
// score[n,k] = ||c_k||^2 - 2 * x_n . c_k   (||x||^2 constant per row -> dropped)
// x.c computed as x_hi*c_hi + x_hi*c_lo + x_lo*c_hi with v_wmma_f32_16x16x32_bf16.
//
// Fast path (needs 516KB workspace): prep kernel converts the codebook to bf16
// hi/lo planes + fp32 ||c||^2 ONCE; the GEMM kernel stages raw bf16 bytes into
// LDS (no per-WG conversion VALU) and prefetches the next block's lines
// (global_prefetch_b8) while computing. Fallback path converts in-kernel.

#define DDIM        128
#define KCODES      1024
#define BLOCK_CODES 128
#define WG_THREADS  256
#define ROWS_PER_WG 128   // 8 waves x 16 rows

typedef __attribute__((ext_vector_type(16))) __bf16          v16bf;
typedef __attribute__((ext_vector_type(8)))  float           v8f;
typedef __attribute__((ext_vector_type(16))) unsigned short  v16u;

union V16 { v16u u; v16bf b; };

__device__ __forceinline__ unsigned short bf16hi(float f) {
  unsigned u = __builtin_bit_cast(unsigned, f);
  u += 0x7FFFu + ((u >> 16) & 1u);       // round-to-nearest-even
  return (unsigned short)(u >> 16);
}
__device__ __forceinline__ float bf16tof(unsigned short h) {
  unsigned u = ((unsigned)h) << 16;
  return __builtin_bit_cast(float, u);
}

// Load one wave's 16-row A panel (16 x 128 fp32) into bf16 hi/lo registers in
// contiguous k order (k = 16*half + s), matching B: per chunk c, half=0 lanes hold K=0..15,
// half=1 lanes hold K=16..31; M = lane&15.
__device__ __forceinline__ void load_a_panel(const float* __restrict__ rp,
                                             int half, v16bf* a_hi, v16bf* a_lo) {
  #pragma unroll
  for (int c = 0; c < 4; ++c) {
    const int d0 = 32 * c + 16 * half;
    V16 hi, lo;
    #pragma unroll
    for (int g = 0; g < 2; ++g) {                       // +0 and +8 K groups
      const float4* p = (const float4*)(rp + d0 + 8 * g);
      float4 f0 = p[0], f1 = p[1];
      float fv[8] = {f0.x, f0.y, f0.z, f0.w, f1.x, f1.y, f1.z, f1.w};
      #pragma unroll
      for (int j = 0; j < 8; ++j) {
        unsigned short h = bf16hi(fv[j]);
        hi.u[8 * g + j] = h;
        lo.u[8 * g + j] = bf16hi(fv[j] - bf16tof(h));
      }
    }
    a_hi[c] = hi.b; a_lo[c] = lo.b;
  }
}

// ---------------- prep: codebook fp32 -> bf16 hi/lo planes + fp32 ||c||^2 ----
__global__ __launch_bounds__(256)
void vq_prep(const float* __restrict__ cb, unsigned short* __restrict__ whi,
             unsigned short* __restrict__ wlo, float* __restrict__ wc2) {
  const int code = blockIdx.x * 8 + (threadIdx.x >> 5);  // 32 lanes per code row
  const int lane = threadIdx.x & 31;
  float4 f = ((const float4*)(cb + (size_t)code * DDIM))[lane];  // d = lane*4..+3
  float fv[4] = {f.x, f.y, f.z, f.w};
  unsigned short h[4], l[4];
  float ss = 0.f;
  #pragma unroll
  for (int j = 0; j < 4; ++j) {
    ss += fv[j] * fv[j];
    h[j] = bf16hi(fv[j]);
    l[j] = bf16hi(fv[j] - bf16tof(h[j]));
  }
  uint2 ph, pl;
  ph.x = (unsigned)h[0] | ((unsigned)h[1] << 16);
  ph.y = (unsigned)h[2] | ((unsigned)h[3] << 16);
  pl.x = (unsigned)l[0] | ((unsigned)l[1] << 16);
  pl.y = (unsigned)l[2] | ((unsigned)l[3] << 16);
  *(uint2*)(whi + (size_t)code * DDIM + lane * 4) = ph;
  *(uint2*)(wlo + (size_t)code * DDIM + lane * 4) = pl;
  #pragma unroll
  for (int m = 16; m >= 1; m >>= 1) ss += __shfl_xor(ss, m, 32);
  if (lane == 0) wc2[code] = ss;
}

// ---------------- fast GEMM+argmin: stages prepacked bf16 into LDS ----------
__global__ __launch_bounds__(WG_THREADS)
void vq_argmin_wmma_pre(const float* __restrict__ x,
                        const unsigned short* __restrict__ whi,
                        const unsigned short* __restrict__ wlo,
                        const float* __restrict__ wc2,
                        int* __restrict__ out)
{
  __shared__ unsigned short s_hi[BLOCK_CODES * DDIM];   // 32 KB
  __shared__ unsigned short s_lo[BLOCK_CODES * DDIM];   // 32 KB
  __shared__ float s_c2[BLOCK_CODES];

  const int tid  = threadIdx.x;
  const int lane = tid & 31;
  const int wave = tid >> 5;
  const int half = lane >> 4;
  const int l15  = lane & 15;
  const int rowBase = blockIdx.x * ROWS_PER_WG + wave * 16;

  v16bf a_hi[4], a_lo[4];
  load_a_panel(x + (size_t)(rowBase + l15) * DDIM, half, a_hi, a_lo);

  float best[8];
  int   bidx[8];
  #pragma unroll
  for (int r = 0; r < 8; ++r) { best[r] = 3.4e38f; bidx[r] = 0; }

  for (int blk = 0; blk < KCODES / BLOCK_CODES; ++blk) {
    // Raw byte staging: 32KB hi + 32KB lo = 2048+2048 uint4, 8 each per thread.
    {
      const uint4* ghi = (const uint4*)(whi + (size_t)blk * BLOCK_CODES * DDIM);
      const uint4* glo = (const uint4*)(wlo + (size_t)blk * BLOCK_CODES * DDIM);
      uint4* shi = (uint4*)s_hi;
      uint4* slo = (uint4*)s_lo;
      #pragma unroll
      for (int i = 0; i < 8; ++i) {
        shi[tid + 256 * i] = ghi[tid + 256 * i];
        slo[tid + 256 * i] = glo[tid + 256 * i];
      }
      if (tid < BLOCK_CODES) s_c2[tid] = wc2[blk * BLOCK_CODES + tid];
    }
    __syncthreads();

    // Prefetch next block's staging lines while this block computes:
    // 2 x 32KB = 512 cachelines of 128B; 256 threads x 2 prefetches covers it.
    if (blk + 1 < KCODES / BLOCK_CODES) {
      const char* nhi = (const char*)(whi + (size_t)(blk + 1) * BLOCK_CODES * DDIM);
      const char* nlo = (const char*)(wlo + (size_t)(blk + 1) * BLOCK_CODES * DDIM);
      __builtin_prefetch(nhi + tid * 128, 0, 3);
      __builtin_prefetch(nlo + tid * 128, 0, 3);
    }

    // Tile pairs: two independent accumulator chains share the A panel.
    #pragma unroll
    for (int t = 0; t < 8; t += 2) {
      const int code0 = t * 16 + l15;
      const int code1 = code0 + 16;
      const float c20 = s_c2[code0];
      const float c21 = s_c2[code1];
      v8f acc0 = {}, acc1 = {};
      #pragma unroll
      for (int c = 0; c < 4; ++c) {
        const int doff = 32 * c + 16 * half;
        v16bf Bh0 = *(const v16bf*)(s_hi + code0 * DDIM + doff);
        v16bf Bl0 = *(const v16bf*)(s_lo + code0 * DDIM + doff);
        v16bf Bh1 = *(const v16bf*)(s_hi + code1 * DDIM + doff);
        v16bf Bl1 = *(const v16bf*)(s_lo + code1 * DDIM + doff);
        acc0 = __builtin_amdgcn_wmma_f32_16x16x32_bf16(false, a_hi[c], false, Bh0,
                                                       (short)0, acc0, false, false);
        acc1 = __builtin_amdgcn_wmma_f32_16x16x32_bf16(false, a_hi[c], false, Bh1,
                                                       (short)0, acc1, false, false);
        acc0 = __builtin_amdgcn_wmma_f32_16x16x32_bf16(false, a_hi[c], false, Bl0,
                                                       (short)0, acc0, false, false);
        acc1 = __builtin_amdgcn_wmma_f32_16x16x32_bf16(false, a_hi[c], false, Bl1,
                                                       (short)0, acc1, false, false);
        acc0 = __builtin_amdgcn_wmma_f32_16x16x32_bf16(false, a_lo[c], false, Bh0,
                                                       (short)0, acc0, false, false);
        acc1 = __builtin_amdgcn_wmma_f32_16x16x32_bf16(false, a_lo[c], false, Bh1,
                                                       (short)0, acc1, false, false);
      }
      const int cg0 = blk * BLOCK_CODES + code0;
      const int cg1 = cg0 + 16;
      #pragma unroll
      for (int r = 0; r < 8; ++r) {
        float s0 = c20 - 2.0f * acc0[r];
        float s1 = c21 - 2.0f * acc1[r];
        if (s0 < best[r]) { best[r] = s0; bidx[r] = cg0; }
        if (s1 < best[r]) { best[r] = s1; bidx[r] = cg1; }
      }
    }
    __syncthreads();
  }

  #pragma unroll
  for (int r = 0; r < 8; ++r) {
    float b = best[r]; int bi = bidx[r];
    #pragma unroll
    for (int m = 8; m >= 1; m >>= 1) {
      float ob = __shfl_xor(b, m, 32);
      int   oi = __shfl_xor(bi, m, 32);
      if (ob < b || (ob == b && oi < bi)) { b = ob; bi = oi; }
    }
    if (l15 == 0) out[rowBase + 8 * half + r] = bi;
  }
}

// ---------------- fallback: self-contained (converts codebook per WG) -------
__global__ __launch_bounds__(WG_THREADS)
void vq_argmin_wmma(const float* __restrict__ x,
                    const float* __restrict__ cb,
                    int* __restrict__ out)
{
  __shared__ unsigned short s_hi[BLOCK_CODES * DDIM];
  __shared__ unsigned short s_lo[BLOCK_CODES * DDIM];
  __shared__ float s_part[2][BLOCK_CODES];
  __shared__ float s_c2[BLOCK_CODES];

  const int tid  = threadIdx.x;
  const int lane = tid & 31;
  const int wave = tid >> 5;
  const int half = lane >> 4;
  const int l15  = lane & 15;
  const int rowBase = blockIdx.x * ROWS_PER_WG + wave * 16;

  v16bf a_hi[4], a_lo[4];
  load_a_panel(x + (size_t)(rowBase + l15) * DDIM, half, a_hi, a_lo);

  float best[8];
  int   bidx[8];
  #pragma unroll
  for (int r = 0; r < 8; ++r) { best[r] = 3.4e38f; bidx[r] = 0; }

  for (int blk = 0; blk < KCODES / BLOCK_CODES; ++blk) {
    {
      const int code_l = tid & 127;
      const int hsel   = tid >> 7;
      const int d0     = hsel * 64;
      const float* cp  = cb + (size_t)(blk * BLOCK_CODES + code_l) * DDIM + d0;
      float ss = 0.f;
      #pragma unroll
      for (int i = 0; i < 16; ++i) {
        float4 f = ((const float4*)cp)[i];
        float fv[4] = {f.x, f.y, f.z, f.w};
        #pragma unroll
        for (int j = 0; j < 4; ++j) {
          float v = fv[j];
          ss += v * v;
          unsigned short h = bf16hi(v);
          const int idx = code_l * DDIM + d0 + 4 * i + j;
          s_hi[idx] = h;
          s_lo[idx] = bf16hi(v - bf16tof(h));
        }
      }
      s_part[hsel][code_l] = ss;
    }
    __syncthreads();
    if (tid < BLOCK_CODES) s_c2[tid] = s_part[0][tid] + s_part[1][tid];
    __syncthreads();

    // Prefetch next block's fp32 codebook lines (64KB = 512 x 128B).
    if (blk + 1 < KCODES / BLOCK_CODES) {
      const char* nxt = (const char*)(cb + (size_t)(blk + 1) * BLOCK_CODES * DDIM);
      __builtin_prefetch(nxt + tid * 128, 0, 3);
      __builtin_prefetch(nxt + 32768 + tid * 128, 0, 3);
    }

    #pragma unroll
    for (int t = 0; t < 8; ++t) {
      const int code_l = t * 16 + l15;
      const float c2v  = s_c2[code_l];
      v8f acc = {};
      #pragma unroll
      for (int c = 0; c < 4; ++c) {
        const int doff = 32 * c + 16 * half;
        v16bf Bh = *(const v16bf*)(s_hi + code_l * DDIM + doff);
        v16bf Bl = *(const v16bf*)(s_lo + code_l * DDIM + doff);
        acc = __builtin_amdgcn_wmma_f32_16x16x32_bf16(false, a_hi[c], false, Bh,
                                                      (short)0, acc, false, false);
        acc = __builtin_amdgcn_wmma_f32_16x16x32_bf16(false, a_hi[c], false, Bl,
                                                      (short)0, acc, false, false);
        acc = __builtin_amdgcn_wmma_f32_16x16x32_bf16(false, a_lo[c], false, Bh,
                                                      (short)0, acc, false, false);
      }
      const int code_g = blk * BLOCK_CODES + code_l;
      #pragma unroll
      for (int r = 0; r < 8; ++r) {
        float s = c2v - 2.0f * acc[r];
        if (s < best[r]) { best[r] = s; bidx[r] = code_g; }
      }
    }
    __syncthreads();
  }

  #pragma unroll
  for (int r = 0; r < 8; ++r) {
    float b = best[r]; int bi = bidx[r];
    #pragma unroll
    for (int m = 8; m >= 1; m >>= 1) {
      float ob = __shfl_xor(b, m, 32);
      int   oi = __shfl_xor(bi, m, 32);
      if (ob < b || (ob == b && oi < bi)) { b = ob; bi = oi; }
    }
    if (l15 == 0) out[rowBase + 8 * half + r] = bi;
  }
}

extern "C" void kernel_launch(void* const* d_in, const int* in_sizes, int n_in,
                              void* d_out, int out_size, void* d_ws, size_t ws_size,
                              hipStream_t stream) {
  (void)n_in; (void)out_size;
  const float* xin = (const float*)d_in[0];   // (16,64,64,128) fp32
  const float* cbk = (const float*)d_in[1];   // (1024,128) fp32
  int* out = (int*)d_out;                     // (65536) int32 argmin indices

  const int rows = in_sizes[0] / DDIM;        // 65536
  const int grid = rows / ROWS_PER_WG;        // 512 workgroups

  const size_t planeElems = (size_t)KCODES * DDIM;            // 131072
  const size_t need = planeElems * 2 * sizeof(unsigned short) // hi + lo
                    + KCODES * sizeof(float);                 // c2  => 516 KB
  if (ws_size >= need) {
    unsigned short* whi = (unsigned short*)d_ws;
    unsigned short* wlo = whi + planeElems;
    float*          wc2 = (float*)(wlo + planeElems);
    vq_prep<<<KCODES / 8, 256, 0, stream>>>(cbk, whi, wlo, wc2);
    vq_argmin_wmma_pre<<<grid, WG_THREADS, 0, stream>>>(xin, whi, wlo, wc2, out);
  } else {
    vq_argmin_wmma<<<grid, WG_THREADS, 0, stream>>>(xin, cbk, out);
  }
}
